// SelfAttentionLayer_3238405341423
// MI455X (gfx1250) — matrix operand order repaired and checked by reading, not compiled, not run on hardware
//
#include <hip/hip_runtime.h>
#include <hip/hip_bf16.h>
#include <stdint.h>

// ---------------------------------------------------------------------------
// MI455X (gfx1250) fused causal self-attention forward.
// bf16 WMMA (v_wmma_f32_16x16x32_bf16) for all GEMMs, fp32 accumulate.
// Global->LDS staging via CDNA5 async-to-LDS DMA (ASYNCcnt).
// B=2, T=2048, D=1024, H=16, hd=64 (fixed by the reference setup).
// ---------------------------------------------------------------------------

typedef unsigned short u16;
typedef __attribute__((ext_vector_type(16))) __bf16 v16bf;
typedef __attribute__((ext_vector_type(8)))  float  v8f;

union Frag {            // 8 VGPRs = 16 bf16 values per lane
    uint32_t u[8];
    v16bf    v;
};

__device__ __forceinline__ v8f wmma_bf16(const Frag& a, const Frag& b, v8f c) {
    // (neg_a, A, neg_b, B, c_mod, C, reuse_a, reuse_b)
    return __builtin_amdgcn_wmma_f32_16x16x32_bf16(false, a.v, false, b.v,
                                                   (short)0, c, false, false);
}

__device__ __forceinline__ u16 f32_to_bf16(float f) {
    uint32_t u = __float_as_uint(f);
    u += 0x7fffu + ((u >> 16) & 1u);       // round-to-nearest-even
    return (u16)(u >> 16);
}

// --- CDNA5 async global->LDS DMA (tracked by ASYNCcnt, not LOADcnt) --------
// VDST operand = 32-bit LDS byte offset (flat-LDS addr[31:0] per aperture calc).
__device__ __forceinline__ void async_ld_b128(void* lds_ptr, const void* gptr) {
    uint32_t loff = (uint32_t)(uintptr_t)lds_ptr;
    asm volatile("global_load_async_to_lds_b128 %0, %1, off"
                 :: "v"(loff), "v"(gptr)
                 : "memory");
}
__device__ __forceinline__ void wait_async0() {
    asm volatile("s_wait_asynccnt 0x0" ::: "memory");
}

// ------------------------------ utility kernels ----------------------------

__global__ void zero_f32_kernel(float* p, size_t n) {
    size_t i = (size_t)blockIdx.x * blockDim.x + threadIdx.x;
    size_t stride = (size_t)gridDim.x * blockDim.x;
    for (; i < n; i += stride) p[i] = 0.0f;
}

__global__ void cvt_bf16_kernel(const float* __restrict__ src, u16* __restrict__ dst, size_t n) {
    size_t i = (size_t)blockIdx.x * blockDim.x + threadIdx.x;
    size_t stride = (size_t)gridDim.x * blockDim.x;
    for (; i < n; i += stride) dst[i] = f32_to_bf16(src[i]);
}

// ------------------------------ LayerNorm ----------------------------------
// One wave32 per token (D=1024 -> 32 elements per lane). Output bf16.

__global__ void ln_kernel(const float* __restrict__ x,
                          const float* __restrict__ lnw,
                          const float* __restrict__ lnb,
                          u16* __restrict__ xn, int D, int ntok) {
    int wave = (blockIdx.x * blockDim.x + threadIdx.x) >> 5;
    int lane = threadIdx.x & 31;
    if (wave >= ntok) return;
    const float* row = x + (size_t)wave * D;
    float s = 0.0f, s2 = 0.0f;
    for (int j = lane; j < D; j += 32) { float v = row[j]; s += v; s2 += v * v; }
    for (int o = 16; o; o >>= 1) { s += __shfl_xor(s, o, 32); s2 += __shfl_xor(s2, o, 32); }
    float mu   = s / (float)D;
    float var  = s2 / (float)D - mu * mu;
    float rstd = rsqrtf(var + 1e-5f);
    u16* orow = xn + (size_t)wave * D;
    for (int j = lane; j < D; j += 32) {
        float v = (row[j] - mu) * rstd * lnw[j] + lnb[j];
        orow[j] = f32_to_bf16(v);
    }
}

// ------------------------------ GEMM helpers -------------------------------
// Block tile 64(M) x 128(N), K-step 32, 256 threads = 8 waves, each wave 32x32.
// A [M,K] row-major bf16, B [N,K] row-major bf16 (NT layout).

#define BM 64
#define BN 128
#define BK 32

// A-fragment K offset for VGPR v, lane-group g (bf16 16x32 layout)
__device__ __forceinline__ int a_koff(int v, int g) {
    return ((v & 4) ? 16 : 0) + 2 * (v & 3) + 8 * g;
}
// B-fragment K offset (bf16 32x16 layout)
__device__ __forceinline__ int b_koff(int v, int g) {
    return ((v & 4) ? 16 : 0) + 2 * (v & 3) + 8 * g;
}

// QKV GEMM: M=4096 (B*T), N=3072 (3D), K=1024. Epilogue adds bias, scatters to
// Q [B,H,T,hd] (pre-scaled by 1/sqrt(hd)), K [B,H,T,hd], V^T [B,H,hd,T], bf16.
__global__ void gemm_qkv_kernel(const u16* __restrict__ A,
                                const u16* __restrict__ Bw,
                                const float* __restrict__ bias,
                                u16* __restrict__ qb, u16* __restrict__ kb,
                                u16* __restrict__ vtb,
                                int K, int T) {
    __shared__ u16 As[BM * BK];
    __shared__ u16 Bs[BN * BK];
    const int m0 = blockIdx.y * BM, n0 = blockIdx.x * BN;
    const int tid = threadIdx.x, lane = tid & 31, wave = tid >> 5;
    const int g = lane >> 4, l16 = lane & 15;
    const int wm = wave >> 2, wn = wave & 3;     // wave -> 32x32 sub-tile
    const int ra = tid >> 2, ca = (tid & 3) * 8;     // A-tile DMA mapping
    const int rb = tid >> 1, cb = (tid & 1) * 16;    // B-tile DMA mapping
    v8f c[2][2] = {};

    for (int kk = 0; kk < K; kk += BK) {
        __syncthreads();                             // prev readers done
        async_ld_b128(&As[ra * BK + ca],
                      &A[(size_t)(m0 + ra) * K + kk + ca]);
        async_ld_b128(&Bs[rb * BK + cb],
                      &Bw[(size_t)(n0 + rb) * K + kk + cb]);
        async_ld_b128(&Bs[rb * BK + cb + 8],
                      &Bw[(size_t)(n0 + rb) * K + kk + cb + 8]);
        if (kk + BK < K) {   // global_prefetch_b8 for the next K tile
            __builtin_prefetch(&A [(size_t)(m0 + ra) * K + kk + BK], 0, 1);
            __builtin_prefetch(&Bw[(size_t)(n0 + rb) * K + kk + BK], 0, 1);
        }
        wait_async0();                               // this wave's DMAs landed
        __syncthreads();                             // all waves' DMAs landed

        Frag a[2], b[2];
#pragma unroll
        for (int mt = 0; mt < 2; ++mt)
#pragma unroll
            for (int v = 0; v < 8; ++v)
                a[mt].u[v] = *(const uint32_t*)&As[(wm * 32 + mt * 16 + l16) * BK + a_koff(v, g)];
#pragma unroll
        for (int nt = 0; nt < 2; ++nt)
#pragma unroll
            for (int v = 0; v < 8; ++v)
                b[nt].u[v] = *(const uint32_t*)&Bs[(wn * 32 + nt * 16 + l16) * BK + b_koff(v, g)];
#pragma unroll
        for (int mt = 0; mt < 2; ++mt)
#pragma unroll
            for (int nt = 0; nt < 2; ++nt)
                c[mt][nt] = wmma_bf16(a[mt], b[nt], c[mt][nt]);
    }

#pragma unroll
    for (int mt = 0; mt < 2; ++mt)
#pragma unroll
        for (int nt = 0; nt < 2; ++nt)
#pragma unroll
            for (int r = 0; r < 8; ++r) {
                int m = m0 + wm * 32 + mt * 16 + r + 8 * g;   // token index
                int e = n0 + wn * 32 + nt * 16 + l16;         // 0..3071
                float val = c[mt][nt][r] + bias[e];
                int bidx = m / T, t = m % T;
                if (e < 1024) {
                    int h = e >> 6, d = e & 63;
                    qb[(((size_t)(bidx * 16 + h)) * T + t) * 64 + d] = f32_to_bf16(val * 0.125f);
                } else if (e < 2048) {
                    int e2 = e - 1024, h = e2 >> 6, d = e2 & 63;
                    kb[(((size_t)(bidx * 16 + h)) * T + t) * 64 + d] = f32_to_bf16(val);
                } else {
                    int e2 = e - 2048, h = e2 >> 6, d = e2 & 63;
                    vtb[(((size_t)(bidx * 16 + h)) * 64 + d) * T + t] = f32_to_bf16(val);
                }
            }
}

// Out-proj GEMM: M=4096, N=1024, K=1024; epilogue: + bias + residual, fp32 out.
__global__ void gemm_out_kernel(const u16* __restrict__ A,
                                const u16* __restrict__ Bw,
                                const float* __restrict__ bias,
                                const float* __restrict__ resid,
                                float* __restrict__ out,
                                int N, int K) {
    __shared__ u16 As[BM * BK];
    __shared__ u16 Bs[BN * BK];
    const int m0 = blockIdx.y * BM, n0 = blockIdx.x * BN;
    const int tid = threadIdx.x, lane = tid & 31, wave = tid >> 5;
    const int g = lane >> 4, l16 = lane & 15;
    const int wm = wave >> 2, wn = wave & 3;
    const int ra = tid >> 2, ca = (tid & 3) * 8;
    const int rb = tid >> 1, cb = (tid & 1) * 16;
    v8f c[2][2] = {};

    for (int kk = 0; kk < K; kk += BK) {
        __syncthreads();
        async_ld_b128(&As[ra * BK + ca],
                      &A[(size_t)(m0 + ra) * K + kk + ca]);
        async_ld_b128(&Bs[rb * BK + cb],
                      &Bw[(size_t)(n0 + rb) * K + kk + cb]);
        async_ld_b128(&Bs[rb * BK + cb + 8],
                      &Bw[(size_t)(n0 + rb) * K + kk + cb + 8]);
        if (kk + BK < K) {
            __builtin_prefetch(&A [(size_t)(m0 + ra) * K + kk + BK], 0, 1);
            __builtin_prefetch(&Bw[(size_t)(n0 + rb) * K + kk + BK], 0, 1);
        }
        wait_async0();
        __syncthreads();

        Frag a[2], b[2];
#pragma unroll
        for (int mt = 0; mt < 2; ++mt)
#pragma unroll
            for (int v = 0; v < 8; ++v)
                a[mt].u[v] = *(const uint32_t*)&As[(wm * 32 + mt * 16 + l16) * BK + a_koff(v, g)];
#pragma unroll
        for (int nt = 0; nt < 2; ++nt)
#pragma unroll
            for (int v = 0; v < 8; ++v)
                b[nt].u[v] = *(const uint32_t*)&Bs[(wn * 32 + nt * 16 + l16) * BK + b_koff(v, g)];
#pragma unroll
        for (int mt = 0; mt < 2; ++mt)
#pragma unroll
            for (int nt = 0; nt < 2; ++nt)
                c[mt][nt] = wmma_bf16(a[mt], b[nt], c[mt][nt]);
    }

#pragma unroll
    for (int mt = 0; mt < 2; ++mt)
#pragma unroll
        for (int nt = 0; nt < 2; ++nt)
#pragma unroll
            for (int r = 0; r < 8; ++r) {
                int m = m0 + wm * 32 + mt * 16 + r + 8 * g;
                int e = n0 + wn * 32 + nt * 16 + l16;
                size_t idx = (size_t)m * N + e;
                out[idx] = c[mt][nt][r] + bias[e] + resid[idx];
            }
}

// ------------------------------ attention ----------------------------------
// One block per (b, h, 32 query rows). Full causal score stripe kept in LDS
// (32 x 2048 fp32 = 256 KB -- only possible in CDNA5's 320 KB WGP LDS).
// Softmax in place, P converted to bf16 in place, P*V via WMMA, and P/H
// atomically accumulated into attn_avg.

__global__ void attn_kernel(const u16* __restrict__ qbf,
                            const u16* __restrict__ kbf,
                            const u16* __restrict__ vtbf,
                            float* __restrict__ attn_avg,
                            u16* __restrict__ ctx,
                            int T) {
    extern __shared__ unsigned char smem[];
    float* S  = (float*)smem;                              // [32][2048] fp32
    u16*   Qs = (u16*)(smem + (size_t)32 * 2048 * 4);      // [32][64]  bf16
    const int H = 16, hd = 64;

    const int blk = blockIdx.x;
    const int nqt = T / 32;
    const int qt = blk % nqt, h = (blk / nqt) % H, b = blk / (nqt * H);
    const int q0 = qt * 32;
    const int klen = q0 + 32;                              // causal truncation
    const int tid = threadIdx.x, lane = tid & 31, wave = tid >> 5;
    const int g = lane >> 4, l16 = lane & 15;

    // --- stage Q tile via async DMA (pre-scaled by 1/sqrt(hd)) ---
    const u16* qbase = qbf + (((size_t)(b * H + h)) * T + q0) * hd;
    { int r = tid >> 3, c = (tid & 7) * 8;
      async_ld_b128(&Qs[r * 64 + c], &qbase[r * 64 + c]); }
    wait_async0();
    __syncthreads();

    // --- scores: S[32][klen] = (Q/8) * K^T, causal-masked ---
    const u16* kbase = kbf + ((size_t)(b * H + h)) * T * hd;
    const int ntiles = klen >> 4;
    for (int nt = wave; nt < ntiles; nt += 8) {
        const int n0 = nt * 16;
        v8f c0 = {}, c1 = {};
#pragma unroll
        for (int ks = 0; ks < 2; ++ks) {                   // hd = 64 = 2 x 32
            Frag a0, a1, bf;
#pragma unroll
            for (int v = 0; v < 8; ++v) {
                int ka = ks * 32 + a_koff(v, g);
                a0.u[v] = *(const uint32_t*)&Qs[l16 * 64 + ka];
                a1.u[v] = *(const uint32_t*)&Qs[(16 + l16) * 64 + ka];
                int kbo = ks * 32 + b_koff(v, g);
                bf.u[v] = *(const uint32_t*)&kbase[(size_t)(n0 + l16) * hd + kbo];
            }
            c0 = wmma_bf16(a0, bf, c0);
            c1 = wmma_bf16(a1, bf, c1);
        }
#pragma unroll
        for (int r = 0; r < 8; ++r) {
            int key = n0 + l16;
            int m = r + 8 * g;
            S[(size_t)m * 2048 + key]        = (key <= q0 + m)      ? c0[r] : -1e30f;
            S[(size_t)(16 + m) * 2048 + key] = (key <= q0 + 16 + m) ? c1[r] : -1e30f;
        }
    }
    __syncthreads();

    // --- softmax per row (4 rows per wave), write P bf16 in place,
    //     accumulate P/H into attn_avg ---
    for (int rr = 0; rr < 4; ++rr) {
        const int row = wave * 4 + rr;
        float* Sr = S + (size_t)row * 2048;
        const int q = q0 + row;
        const int len = q + 1;
        float mx = -1e30f;
        for (int j = lane; j < len; j += 32) mx = fmaxf(mx, Sr[j]);
        for (int o = 16; o; o >>= 1) mx = fmaxf(mx, __shfl_xor(mx, o, 32));
        float sum = 0.0f;
        for (int j = lane; j < len; j += 32) { float e = __expf(Sr[j] - mx); Sr[j] = e; sum += e; }
        for (int o = 16; o; o >>= 1) sum += __shfl_xor(sum, o, 32);
        const float inv = 1.0f / sum;
        float* avg_row = attn_avg + ((size_t)b * T + q) * T;
        u16* Pr = (u16*)Sr;
        const int chunks = klen >> 5;                       // klen % 32 == 0
        for (int cch = 0; cch < chunks; ++cch) {
            int j = cch * 32 + lane;
            float p = (j < len) ? Sr[j] * inv : 0.0f;       // read-before-write safe:
            atomicAdd(&avg_row[j], p * (1.0f / 16.0f));     // bf16 word j/2 < next reads
            Pr[j] = f32_to_bf16(p);
        }
    }
    __syncthreads();

    // --- P (bf16, in LDS) x V^T -> ctx bf16 [B,T,D] ---
    const u16* vtb = vtbf + ((size_t)(b * H + h)) * hd * T;
    const int mt = wave >> 2, ntl = wave & 3;              // 2 x 4 wave grid
    const u16* Prow = (const u16*)S + (size_t)(mt * 16 + l16) * 4096;
    const u16* Vrow = vtb + (size_t)(ntl * 16 + l16) * T;
    v8f acc = {};
    for (int k0 = 0; k0 < klen; k0 += 32) {
        Frag a, bf;
#pragma unroll
        for (int v = 0; v < 8; ++v) {
            a.u[v]  = *(const uint32_t*)&Prow[k0 + a_koff(v, g)];
            bf.u[v] = *(const uint32_t*)&Vrow[k0 + b_koff(v, g)];
        }
        acc = wmma_bf16(a, bf, acc);
    }
#pragma unroll
    for (int r = 0; r < 8; ++r) {
        int qloc = mt * 16 + r + 8 * g;
        int t = q0 + qloc;
        int dcol = h * 64 + ntl * 16 + l16;
        ctx[((size_t)(b * T) + t) * 1024 + dcol] = f32_to_bf16(acc[r]);
    }
}

// ------------------------------ launch -------------------------------------

extern "C" void kernel_launch(void* const* d_in, const int* in_sizes, int n_in,
                              void* d_out, int out_size, void* d_ws, size_t ws_size,
                              hipStream_t stream) {
    const float* x     = (const float*)d_in[0];
    const float* w_in  = (const float*)d_in[1];
    const float* b_in  = (const float*)d_in[2];
    const float* w_out = (const float*)d_in[3];
    const float* b_out = (const float*)d_in[4];
    const float* lnw   = (const float*)d_in[5];
    const float* lnb   = (const float*)d_in[6];

    constexpr int Bb = 2, T = 2048, D = 1024, H = 16;
    constexpr int M = Bb * T;                       // 4096 tokens

    float* out      = (float*)d_out;                // [B,T,D]
    float* attn_avg = out + (size_t)M * D;          // [B,T,T]

    char* ws = (char*)d_ws;
    u16* xn   = (u16*)ws; ws += (size_t)M * D * 2;          // LN(x) bf16
    u16* wqkv = (u16*)ws; ws += (size_t)3 * D * D * 2;      // in_proj_w bf16
    u16* wo   = (u16*)ws; ws += (size_t)D * D * 2;          // out_proj_w bf16
    u16* qb   = (u16*)ws; ws += (size_t)M * D * 2;          // Q  [B,H,T,hd]
    u16* kbuf = (u16*)ws; ws += (size_t)M * D * 2;          // K  [B,H,T,hd]
    u16* vtb  = (u16*)ws; ws += (size_t)M * D * 2;          // V^T[B,H,hd,T]
    u16* ctx  = (u16*)ws; ws += (size_t)M * D * 2;          // attn output bf16

    zero_f32_kernel<<<4096, 256, 0, stream>>>(attn_avg, (size_t)Bb * T * T);
    cvt_bf16_kernel<<<3072, 256, 0, stream>>>(w_in,  wqkv, (size_t)3 * D * D);
    cvt_bf16_kernel<<<1024, 256, 0, stream>>>(w_out, wo,   (size_t)D * D);
    ln_kernel<<<M / 8, 256, 0, stream>>>(x, lnw, lnb, xn, D, M);

    gemm_qkv_kernel<<<dim3(3 * D / BN, M / BM), 256, 0, stream>>>(
        xn, wqkv, b_in, qb, kbuf, vtb, D, T);

    size_t attn_smem = (size_t)32 * 2048 * 4 + (size_t)32 * 64 * 2;   // 260 KB
    attn_kernel<<<Bb * H * (T / 32), 256, attn_smem, stream>>>(
        qb, kbuf, vtb, attn_avg, ctx, T);

    gemm_out_kernel<<<dim3(D / BN, M / BM), 256, 0, stream>>>(
        ctx, wo, b_out, x, out, D, D);
}
